// PASSAGE_55121610277322
// MI455X (gfx1250) — matrix operand order repaired and checked by reading, not compiled, not run on hardware
//
#include <hip/hip_runtime.h>
#include <math.h>

#define N_NODES 20000
#define N_EDGES 200000
#define E_TOT   220000
#define NFEAT   2048
#define H1DIM   512
#define H2DIM   30

#define BN      64     // block N tile
#define KSTEP   32     // K per WMMA
#define LDS_ROW 40     // ushorts per staged B row (32 data + 8 pad -> 80B stride)

typedef __attribute__((ext_vector_type(16))) __bf16 v16bf;
typedef __attribute__((ext_vector_type(8)))  float  v8f;

// gfx1250 async global->LDS staging (ASYNCcnt-tracked, no VGPR round-trip).
#if defined(__HIP_DEVICE_COMPILE__) && __has_builtin(__builtin_amdgcn_global_load_async_to_lds_b128)
#define HAVE_ASYNC_LDS 1
// Exact parameter types per hipcc diagnostic:
//   arg0: int __attribute__((vector_size(16))) __device__(AS1)*   (non-const)
//   arg1: LDS-side pointer (AS3)
typedef int v4i_vs __attribute__((vector_size(4 * sizeof(int))));
typedef __attribute__((address_space(1))) v4i_vs* gasv4i;
typedef __attribute__((address_space(3))) v4i_vs* ldsv4i;
#else
#define HAVE_ASYNC_LDS 0
#endif

#if defined(__HIP_DEVICE_COMPILE__)
__device__ inline void wait_asynccnt0() {
#if __has_builtin(__builtin_amdgcn_s_wait_asynccnt)
    __builtin_amdgcn_s_wait_asynccnt(0);
#else
    asm volatile("s_wait_asynccnt 0x0" ::: "memory");
#endif
}
#else
__device__ inline void wait_asynccnt0() {}
#endif

union BFrag {
    v16bf v;
    __bf16 h[16];
    uint4  q[2];
    unsigned short u[16];
};

__device__ inline unsigned short f2bf16(float f) {
    union { __bf16 h; unsigned short u; } cv;
    cv.h = (__bf16)f;
    return cv.u;
}

// ---------------------------------------------------------------------------
// Weight conversion: W1 [2048,512] f32 -> bf16 row-major copy (for h3 @ W1^T)
// and bf16 transposed copy [512,2048] (for features @ W1).
// ---------------------------------------------------------------------------
__global__ void convert_weights(const float* __restrict__ W1,
                                unsigned short* __restrict__ w1b,
                                unsigned short* __restrict__ w1t) {
    int i = blockIdx.x * 256 + threadIdx.x;
    if (i >= NFEAT * H1DIM) return;
    int k = i / H1DIM;        // 0..2047
    int n = i - k * H1DIM;    // 0..511
    unsigned short b = f2bf16(W1[i]);
    w1b[i] = b;                                  // (W1^T)(k,n) = W1[n][k] layout
    w1t[(size_t)n * NFEAT + k] = b;              // W1(k,n) contiguous-in-k layout
}

// ---------------------------------------------------------------------------
// bf16 WMMA GEMM: C[M x N] = A(f32)[M x K] * B[K x N], with B supplied
// pre-transposed in bf16 so element B(k,n) = Bt[n*ldbt + k].
// Block = 256 threads = 8 waves. Block tile 128(M) x 64(N); each wave owns a
// 16x64 strip = 4 accumulators fed by one shared A fragment. The 64x32 B tile
// is staged through double-buffered LDS; on gfx1250 the staging uses
// GLOBAL_LOAD_ASYNC_TO_LDS_B128 (ASYNCcnt) overlapped with the WMMAs, with
// s_wait_asynccnt 0 immediately before the end-of-iteration barrier.
// FUSE_MSE: instead of storing C, accumulate sum((Fref - C)^2) into lossAcc.
// ---------------------------------------------------------------------------
template <bool FUSE_MSE>
__global__ __launch_bounds__(256)
void gemm_bf16_wmma(const float* __restrict__ A, int lda,
                    const unsigned short* __restrict__ Bt, int ldbt,
                    int K, int M, int ldc,
                    float* __restrict__ C,
                    const float* __restrict__ Fref,
                    float* __restrict__ lossAcc) {
    __shared__ unsigned short bstage[2][BN * LDS_ROW];

    const int tid  = threadIdx.x;
    const int lane = tid & 31;
    const int wave = tid >> 5;
    const int hlf  = lane >> 4;       // 0: lanes 0-15, 1: lanes 16-31
    const int lr   = lane & 15;

    const int nBase = blockIdx.x * BN;
    const int mTile = (blockIdx.y * 8 + wave) * 16;
    const bool activeM = (mTile + 16) <= M;           // M % 16 == 0 -> all-or-nothing per wave
    const int  mrow = activeM ? (mTile + lr) : lr;    // clamped row for safe loads

    // A fragment base (contiguous K, matching B):
    //  lanes 0-15 : h[0..15] = K 0..15
    //  lanes 16-31: h[0..15] = K 16..31
    const float* arow = A + (size_t)mrow * lda + hlf * 16;

    // B staging role: thread copies one 16B chunk of the 64x32 bf16 tile.
    const int srow = tid >> 2;                        // 0..63  (n within tile)
    const int schk = (tid & 3) * 8;                   // k offset in ushorts
    const unsigned short* gsrc = Bt + (size_t)(nBase + srow) * ldbt + schk;
    const int soff = srow * LDS_ROW + schk;

    // prologue: stage K tile 0 into buffer 0
#if HAVE_ASYNC_LDS
    __builtin_amdgcn_global_load_async_to_lds_b128(
        (gasv4i)gsrc, (ldsv4i)&bstage[0][soff], 0, 0);
    wait_asynccnt0();
#else
    {
        uint4 g = *reinterpret_cast<const uint4*>(gsrc);
        *reinterpret_cast<uint4*>(&bstage[0][soff]) = g;
    }
#endif
    __syncthreads();

    v8f c0 = {}, c1 = {}, c2 = {}, c3 = {};
    const int nIter = K / KSTEP;
    for (int it = 0; it < nIter; ++it) {
        const int k0  = it * KSTEP;
        const int cur = it & 1;
        const int nxt = cur ^ 1;

        // stage next K tile into the other buffer (safe: it was fully read
        // before the barrier that ended the previous iteration)
        const bool hasNext = (it + 1 < nIter);
        if (hasNext) {
#if HAVE_ASYNC_LDS
            __builtin_amdgcn_global_load_async_to_lds_b128(
                (gasv4i)(gsrc + (size_t)(k0 + KSTEP)),
                (ldsv4i)&bstage[nxt][soff], 0, 0);
#else
            uint4 g = *reinterpret_cast<const uint4*>(gsrc + (size_t)(k0 + KSTEP));
            *reinterpret_cast<uint4*>(&bstage[nxt][soff]) = g;
#endif
            __builtin_prefetch((const void*)(gsrc + (size_t)(k0 + 2 * KSTEP)), 0, 1);
        }

        // A fragment (f32 -> bf16 on the fly; compiler packs to v_cvt_pk_bf16_f32)
        const float4* ap0 = reinterpret_cast<const float4*>(arow + k0);
        float4 a0 = ap0[0];
        float4 a1 = ap0[1];
        const float4* ap1 = reinterpret_cast<const float4*>(arow + k0 + 8);
        float4 a2 = ap1[0];
        float4 a3 = ap1[1];
        BFrag af;
        af.h[0]  = (__bf16)a0.x; af.h[1]  = (__bf16)a0.y;
        af.h[2]  = (__bf16)a0.z; af.h[3]  = (__bf16)a0.w;
        af.h[4]  = (__bf16)a1.x; af.h[5]  = (__bf16)a1.y;
        af.h[6]  = (__bf16)a1.z; af.h[7]  = (__bf16)a1.w;
        af.h[8]  = (__bf16)a2.x; af.h[9]  = (__bf16)a2.y;
        af.h[10] = (__bf16)a2.z; af.h[11] = (__bf16)a2.w;
        af.h[12] = (__bf16)a3.x; af.h[13] = (__bf16)a3.y;
        af.h[14] = (__bf16)a3.z; af.h[15] = (__bf16)a3.w;

        // 4 B fragments from LDS (row n = nt*16 + lr; lanes 0-15 hold K 0..15,
        // lanes 16-31 hold K 16..31 -> 32B = 2 x ds b128)
        const unsigned short* lb0 = &bstage[cur][(0 * 16 + lr) * LDS_ROW + hlf * 16];
        const unsigned short* lb1 = &bstage[cur][(1 * 16 + lr) * LDS_ROW + hlf * 16];
        const unsigned short* lb2 = &bstage[cur][(2 * 16 + lr) * LDS_ROW + hlf * 16];
        const unsigned short* lb3 = &bstage[cur][(3 * 16 + lr) * LDS_ROW + hlf * 16];
        BFrag b0, b1, b2, b3;
        b0.q[0] = reinterpret_cast<const uint4*>(lb0)[0];
        b0.q[1] = reinterpret_cast<const uint4*>(lb0)[1];
        b1.q[0] = reinterpret_cast<const uint4*>(lb1)[0];
        b1.q[1] = reinterpret_cast<const uint4*>(lb1)[1];
        b2.q[0] = reinterpret_cast<const uint4*>(lb2)[0];
        b2.q[1] = reinterpret_cast<const uint4*>(lb2)[1];
        b3.q[0] = reinterpret_cast<const uint4*>(lb3)[0];
        b3.q[1] = reinterpret_cast<const uint4*>(lb3)[1];

        c0 = __builtin_amdgcn_wmma_f32_16x16x32_bf16(false, af.v, false, b0.v,
                                                     (short)0, c0, false, false);
        c1 = __builtin_amdgcn_wmma_f32_16x16x32_bf16(false, af.v, false, b1.v,
                                                     (short)0, c1, false, false);
        c2 = __builtin_amdgcn_wmma_f32_16x16x32_bf16(false, af.v, false, b2.v,
                                                     (short)0, c2, false, false);
        c3 = __builtin_amdgcn_wmma_f32_16x16x32_bf16(false, af.v, false, b3.v,
                                                     (short)0, c3, false, false);

#if HAVE_ASYNC_LDS
        if (hasNext) wait_asynccnt0();   // staging done before the barrier
#endif
        __syncthreads();
    }

    // C/D layout: VGPR r, lanes 0-15 -> (M = r, N = lane); lanes 16-31 -> (M = r+8).
    if (!FUSE_MSE) {
        if (activeM) {
#pragma unroll
            for (int r = 0; r < 8; ++r) {
                int m = mTile + hlf * 8 + r;
                float* crow = C + (size_t)m * ldc + nBase + lr;
                crow[0]  = c0[r];
                crow[16] = c1[r];
                crow[32] = c2[r];
                crow[48] = c3[r];
            }
        }
    } else {
        float acc = 0.f;
        if (activeM) {
#pragma unroll
            for (int r = 0; r < 8; ++r) {
                int m = mTile + hlf * 8 + r;
                const float* frow = Fref + (size_t)m * ldc + nBase + lr;
                float d0 = frow[0]  - c0[r];
                float d1 = frow[16] - c1[r];
                float d2 = frow[32] - c2[r];
                float d3 = frow[48] - c3[r];
                acc += d0 * d0 + d1 * d1 + d2 * d2 + d3 * d3;
            }
        }
#pragma unroll
        for (int off = 16; off > 0; off >>= 1)
            acc += __shfl_xor(acc, off, 32);
        if (lane == 0 && activeM) atomicAdd(lossAcc, acc);
    }
}

// ---------------------------------------------------------------------------
// Attention logits: a1s[n] = xw1[n,:] . att_src ; a1d[n] = xw1[n,:] . att_dst
// One wave per node.
// ---------------------------------------------------------------------------
__global__ __launch_bounds__(256)
void att_logits(const float* __restrict__ xw,
                const float* __restrict__ asrc, const float* __restrict__ adst,
                float* __restrict__ a1s, float* __restrict__ a1d) {
    int node = blockIdx.x * 8 + (threadIdx.x >> 5);
    if (node >= N_NODES) return;
    int lane = threadIdx.x & 31;
    const float* row = xw + (size_t)node * H1DIM;
    float ss = 0.f, sd = 0.f;
    for (int j = lane; j < H1DIM; j += 32) {
        float v = row[j];
        ss += v * asrc[j];
        sd += v * adst[j];
    }
#pragma unroll
    for (int off = 16; off > 0; off >>= 1) {
        ss += __shfl_xor(ss, off, 32);
        sd += __shfl_xor(sd, off, 32);
    }
    if (lane == 0) { a1s[node] = ss; a1d[node] = sd; }
}

// ---------------------------------------------------------------------------
// Segment softmax pass 1: e = sigmoid(a1s[src]+a1d[dst]); per-dst max via
// uint atomicMax (all e > 0, so memset-0 init is a valid identity).
// ---------------------------------------------------------------------------
__global__ void edge_pass1(const int* __restrict__ ei,
                           const float* __restrict__ a1s, const float* __restrict__ a1d,
                           float* __restrict__ esig, unsigned* __restrict__ nmax) {
    int e = blockIdx.x * 256 + threadIdx.x;
    if (e >= E_TOT) return;
    int s, d;
    if (e < N_EDGES) { s = ei[e]; d = ei[N_EDGES + e]; }
    else             { s = d = e - N_EDGES; }           // self loops
    float x = a1s[s] + a1d[d];
    float sig = 1.f / (1.f + __expf(-x));
    esig[e] = sig;
    atomicMax(nmax + d, __float_as_uint(sig));          // order-preserving for >0
}

// Pass 2: ex = exp(e - max[dst]); per-dst sum.
__global__ void edge_pass2(const int* __restrict__ ei,
                           const float* __restrict__ esig,
                           const unsigned* __restrict__ nmax,
                           float* __restrict__ eex, float* __restrict__ nsum) {
    int e = blockIdx.x * 256 + threadIdx.x;
    if (e >= E_TOT) return;
    int d = (e < N_EDGES) ? ei[N_EDGES + e] : (e - N_EDGES);
    float ex = __expf(esig[e] - __uint_as_float(nmax[d]));
    eex[e] = ex;
    atomicAdd(nsum + d, ex);
}

// Message scatter: H[dst,:] += coef * X[src,:], one wave per edge, float4 gather.
__global__ __launch_bounds__(256)
void edge_scatter(const int* __restrict__ ei,
                  const float* __restrict__ eex, const float* __restrict__ nsum,
                  const float* __restrict__ X, float* __restrict__ H) {
    int e = blockIdx.x * 8 + (threadIdx.x >> 5);
    if (e >= E_TOT) return;
    int lane = threadIdx.x & 31;
    int s, d;
    if (e < N_EDGES) { s = ei[e]; d = ei[N_EDGES + e]; }
    else             { s = d = e - N_EDGES; }
    float coef = eex[e] / nsum[d];
    const float4* xs = reinterpret_cast<const float4*>(X + (size_t)s * H1DIM);
    float* hd = H + (size_t)d * H1DIM;
#pragma unroll
    for (int j = lane; j < H1DIM / 4; j += 32) {
        float4 v = xs[j];
        int c = j * 4;
        atomicAdd(hd + c + 0, v.x * coef);
        atomicAdd(hd + c + 1, v.y * coef);
        atomicAdd(hd + c + 2, v.z * coef);
        atomicAdd(hd + c + 3, v.w * coef);
    }
}

// ELU in place.
__global__ void elu_inplace(float* __restrict__ H, int n) {
    int i = blockIdx.x * 256 + threadIdx.x;
    if (i >= n) return;
    float v = H[i];
    H[i] = v > 0.f ? v : (__expf(v) - 1.f);
}

// h2 = normalize_rows(h1 @ W2). One wave per node; lanes 0..29 own a column.
__global__ __launch_bounds__(256)
void h2_norm(const float* __restrict__ H1buf, const float* __restrict__ W2,
             float* __restrict__ H2buf) {
    int node = blockIdx.x * 8 + (threadIdx.x >> 5);
    if (node >= N_NODES) return;
    int lane = threadIdx.x & 31;
    float acc = 0.f;
    if (lane < H2DIM) {
        const float* h = H1buf + (size_t)node * H1DIM;
        for (int k = 0; k < H1DIM; ++k)
            acc += h[k] * W2[k * H2DIM + lane];   // W2 [512,30], coalesced in lane
    }
    float n2 = acc * acc;
#pragma unroll
    for (int off = 16; off > 0; off >>= 1)
        n2 += __shfl_xor(n2, off, 32);
    float norm = fmaxf(sqrtf(n2), 1e-12f);
    if (lane < H2DIM) H2buf[(size_t)node * H2DIM + lane] = acc / norm;
}

// xw3 = h2 @ W2^T : xw3[n,c] = sum_k h2[n,k] * W2[c,k]. Block = one node row.
__global__ __launch_bounds__(512)
void xw3_kernel(const float* __restrict__ H2buf, const float* __restrict__ W2,
                float* __restrict__ X3) {
    __shared__ float hrow[H2DIM];
    int node = blockIdx.x;
    int c = threadIdx.x;                       // 0..511
    if (c < H2DIM) hrow[c] = H2buf[(size_t)node * H2DIM + c];
    __syncthreads();
    float acc = 0.f;
    const float* w = W2 + (size_t)c * H2DIM;   // row c of W2 [512,30]
#pragma unroll
    for (int k = 0; k < H2DIM; ++k) acc += hrow[k] * w[k];
    X3[(size_t)node * H1DIM + c] = acc;
}

__global__ void finalize(const float* __restrict__ lossAcc, float* __restrict__ out) {
    out[0] = lossAcc[0] * (1.0f / ((float)N_NODES * (float)NFEAT));
}

// ---------------------------------------------------------------------------
extern "C" void kernel_launch(void* const* d_in, const int* in_sizes, int n_in,
                              void* d_out, int out_size, void* d_ws, size_t ws_size,
                              hipStream_t stream) {
    (void)in_sizes; (void)n_in; (void)out_size; (void)ws_size;

    const float* features = (const float*)d_in[0];
    const int*   edges    = (const int*)d_in[1];
    const float* W1       = (const float*)d_in[2];
    const float* att_src  = (const float*)d_in[3];
    const float* att_dst  = (const float*)d_in[4];
    const float* W2       = (const float*)d_in[5];
    float* out = (float*)d_out;

    char* base = (char*)d_ws;
    size_t off = 0;
    auto alloc = [&](size_t bytes) -> char* {
        char* p = base + off;
        off += (bytes + 255) & ~(size_t)255;
        return p;
    };
    float*          bufA = (float*)alloc((size_t)N_NODES * H1DIM * 4);   // xw1, later xw3
    float*          bufB = (float*)alloc((size_t)N_NODES * H1DIM * 4);   // h1,  later h3
    unsigned short* w1t  = (unsigned short*)alloc((size_t)H1DIM * NFEAT * 2);
    unsigned short* w1b  = (unsigned short*)alloc((size_t)NFEAT * H1DIM * 2);
    float*          a1s  = (float*)alloc((size_t)N_NODES * 4);
    float*          a1d  = (float*)alloc((size_t)N_NODES * 4);
    unsigned*       nmax = (unsigned*)alloc((size_t)N_NODES * 4);
    float*          nsum = (float*)alloc((size_t)N_NODES * 4);
    float*          esig = (float*)alloc((size_t)E_TOT * 4);
    float*          eex  = (float*)alloc((size_t)E_TOT * 4);
    float*          h2b  = (float*)alloc((size_t)N_NODES * H2DIM * 4);
    float*          loss = (float*)alloc(256);

    const int mWaves = N_NODES / 16;                 // 1250
    const int gridY  = (mWaves + 7) / 8;             // 157

    // 0) weights -> bf16 (both layouts)
    convert_weights<<<(NFEAT * H1DIM + 255) / 256, 256, 0, stream>>>(W1, w1b, w1t);

    // 1) xw1 = features @ W1   (M=20000, N=512, K=2048)
    {
        dim3 g(H1DIM / BN, gridY);
        gemm_bf16_wmma<false><<<g, 256, 0, stream>>>(features, NFEAT, w1t, NFEAT,
                                                     NFEAT, N_NODES, H1DIM,
                                                     bufA, nullptr, nullptr);
    }

    // 2) attention logits
    att_logits<<<N_NODES / 8, 256, 0, stream>>>(bufA, att_src, att_dst, a1s, a1d);

    // 3) segment softmax (coefficients shared by conv1 and conv3)
    (void)hipMemsetAsync(nmax, 0, (size_t)N_NODES * 4, stream);
    (void)hipMemsetAsync(nsum, 0, (size_t)N_NODES * 4, stream);
    (void)hipMemsetAsync(loss, 0, 256, stream);
    edge_pass1<<<(E_TOT + 255) / 256, 256, 0, stream>>>(edges, a1s, a1d, esig, nmax);
    edge_pass2<<<(E_TOT + 255) / 256, 256, 0, stream>>>(edges, esig, nmax, eex, nsum);

    // 4) h1 = elu(scatter(xw1))
    (void)hipMemsetAsync(bufB, 0, (size_t)N_NODES * H1DIM * 4, stream);
    edge_scatter<<<(E_TOT + 7) / 8, 256, 0, stream>>>(edges, eex, nsum, bufA, bufB);
    elu_inplace<<<(N_NODES * H1DIM + 255) / 256, 256, 0, stream>>>(bufB, N_NODES * H1DIM);

    // 5) h2 = normalize(h1 @ W2) ; xw3 = h2 @ W2^T  (overwrites bufA)
    h2_norm<<<N_NODES / 8, 256, 0, stream>>>(bufB, W2, h2b);
    xw3_kernel<<<N_NODES, 512, 0, stream>>>(h2b, W2, bufA);

    // 6) h3 = elu(scatter(xw3))  (tied attention -> reuse eex/nsum)
    (void)hipMemsetAsync(bufB, 0, (size_t)N_NODES * H1DIM * 4, stream);
    edge_scatter<<<(E_TOT + 7) / 8, 256, 0, stream>>>(edges, eex, nsum, bufA, bufB);
    elu_inplace<<<(N_NODES * H1DIM + 255) / 256, 256, 0, stream>>>(bufB, N_NODES * H1DIM);

    // 7) loss = mean((features - h3 @ W1^T)^2), fused epilogue (M=20000,N=2048,K=512)
    {
        dim3 g(NFEAT / BN, gridY);
        gemm_bf16_wmma<true><<<g, 256, 0, stream>>>(bufB, H1DIM, w1b, H1DIM,
                                                    H1DIM, N_NODES, NFEAT,
                                                    nullptr, features, loss);
    }

    finalize<<<1, 1, 0, stream>>>(loss, out);
}
